// QuantizedLinear_73392401154437
// MI455X (gfx1250) — matrix operand order repaired and checked by reading, not compiled, not run on hardware
//
#include <hip/hip_runtime.h>

typedef __attribute__((ext_vector_type(16))) __bf16 v16bf;
typedef __attribute__((ext_vector_type(8)))  __bf16 v8bf;
typedef __attribute__((ext_vector_type(4)))  __bf16 v4bf;
typedef __attribute__((ext_vector_type(8)))  float  v8f;
typedef __attribute__((ext_vector_type(4)))  unsigned int v4u;
typedef __attribute__((ext_vector_type(8)))  int v8i;
typedef __attribute__((ext_vector_type(4)))  int v4i;

#define K_DIM 2048
#define N_DIM 2048
#define BM 128
#define BN 128
#define BK32 32
#define BK64 64

#if defined(__has_builtin)
#  if __has_builtin(__builtin_amdgcn_tensor_load_to_lds)
#    define USE_TDM 1
#  endif
#endif
#ifndef USE_TDM
#  define USE_TDM 0
#endif

#if USE_TDM
// 2D TDM descriptor (CDNA5 ISA) -> tensor_load_to_lds.
__device__ __forceinline__ void tdm_load_2d(unsigned long long gaddr, unsigned lds_off,
                                            unsigned data_size_code,   // 0:1B 1:2B 2:4B
                                            unsigned tensor_d0, unsigned tensor_d1,
                                            unsigned long long d0_stride,
                                            unsigned tile_d0, unsigned tile_d1) {
  v4u g0;
  g0[0] = 1u;                                                     // count=1
  g0[1] = lds_off;                                                // lds_addr
  g0[2] = (unsigned)(gaddr & 0xFFFFFFFFull);                      // global_addr lo
  g0[3] = (unsigned)((gaddr >> 32) & 0x01FFFFFFull) | (2u << 30); // addr hi | type=2
  v8i g1;
  g1[0] = (int)(data_size_code << 16);
  g1[1] = (int)((tensor_d0 & 0xFFFFu) << 16);
  g1[2] = (int)((tensor_d0 >> 16) | ((tensor_d1 & 0xFFFFu) << 16));
  g1[3] = (int)((tensor_d1 >> 16) | (tile_d0 << 16));
  g1[4] = (int)tile_d1;
  g1[5] = (int)(d0_stride & 0xFFFFFFFFull);
  g1[6] = (int)((d0_stride >> 32) & 0xFFFFull);
  g1[7] = 0;
  v4i gz = {0, 0, 0, 0};
#if __clang_major__ >= 23
  v8i gz8 = {0, 0, 0, 0, 0, 0, 0, 0};
  __builtin_amdgcn_tensor_load_to_lds(g0, g1, gz, gz, gz8, 0);
#else
  __builtin_amdgcn_tensor_load_to_lds(g0, g1, gz, gz, 0);
#endif
}
#endif

// ---------------- f32 -> bf16 bulk convert (used for w, and x on fast path) ----
__global__ __launch_bounds__(256) void bitnet_cvt(const float* __restrict__ src,
                                                  __bf16* __restrict__ dst, int n4) {
  int i = blockIdx.x * 256 + threadIdx.x;
  if (i < n4) {
    float4 f = reinterpret_cast<const float4*>(src)[i];
    v4bf o;
    o[0] = (__bf16)f.x; o[1] = (__bf16)f.y; o[2] = (__bf16)f.z; o[3] = (__bf16)f.w;
    reinterpret_cast<v4bf*>(dst)[i] = o;
  }
}

// ---------------- Fast path: all-bf16 WMMA GEMM, BK=64 double-buffered --------
__global__ __launch_bounds__(256) void bitnet_gemm_bf16(const __bf16* __restrict__ xbf,
                                                        const float* __restrict__ input_scale,
                                                        const __bf16* __restrict__ wbf,
                                                        const float* __restrict__ weight_scale,
                                                        float* __restrict__ out,
                                                        float* __restrict__ partials) {
  __shared__ __align__(32) __bf16 sA[2][BM * BK64];   // 2 x 16 KB
  __shared__ __align__(32) __bf16 sB[2][BN * BK64];   // 2 x 16 KB

  const int tid     = threadIdx.x;
  const int lane    = tid & 31;
  const int wave    = __builtin_amdgcn_readfirstlane(tid >> 5);
  const int waveM   = wave & 3;
  const int waveN   = wave >> 2;
  const int lrow    = lane & 15;
  const int halfsel = lane >> 4;

  const int blockM = blockIdx.y * BM;
  const int blockN = blockIdx.x * BN;

  v8f zero = {0.f, 0.f, 0.f, 0.f, 0.f, 0.f, 0.f, 0.f};
  v8f acc[2][4];
#pragma unroll
  for (int mt = 0; mt < 2; ++mt)
#pragma unroll
    for (int nt = 0; nt < 4; ++nt) acc[mt][nt] = zero;

  const unsigned long long xbase = (unsigned long long)(const void*)(xbf + (size_t)blockM * K_DIM);
  const unsigned long long wbase = (unsigned long long)(const void*)(wbf + (size_t)blockN * K_DIM);

#if USE_TDM
  const unsigned ldsA[2] = { (unsigned)(unsigned long long)(uintptr_t)&sA[0][0],
                             (unsigned)(unsigned long long)(uintptr_t)&sA[1][0] };
  const unsigned ldsB[2] = { (unsigned)(unsigned long long)(uintptr_t)&sB[0][0],
                             (unsigned)(unsigned long long)(uintptr_t)&sB[1][0] };
  if (wave == 0) {
    tdm_load_2d(xbase, ldsA[0], 1u, K_DIM, BM, K_DIM, BK64, BM);
    tdm_load_2d(wbase, ldsB[0], 1u, K_DIM, BN, K_DIM, BK64, BN);
    __builtin_amdgcn_s_wait_tensorcnt(0);
  }
  __syncthreads();
#else
  for (int i = tid; i < BM * BK64 / 8; i += 256) {
    int r = (i * 8) / BK64, k = (i * 8) % BK64;
    reinterpret_cast<uint4*>(&sA[0][0])[i] =
        *reinterpret_cast<const uint4*>((const __bf16*)xbase + (size_t)r * K_DIM + k);
  }
  for (int i = tid; i < BN * BK64 / 8; i += 256) {
    int r = (i * 8) / BK64, k = (i * 8) % BK64;
    reinterpret_cast<uint4*>(&sB[0][0])[i] =
        *reinterpret_cast<const uint4*>((const __bf16*)wbase + (size_t)r * K_DIM + k);
  }
  __syncthreads();
#endif

#pragma unroll 1
  for (int kt = 0; kt < K_DIM / BK64; ++kt) {
    const int buf = kt & 1;

    if (kt + 1 < K_DIM / BK64) {
      const unsigned long long k0 = (unsigned long long)(kt + 1) * BK64;
#if USE_TDM
      if (wave == 0) {
        tdm_load_2d(xbase + k0 * 2, ldsA[buf ^ 1], 1u, K_DIM, BM, K_DIM, BK64, BM);
        tdm_load_2d(wbase + k0 * 2, ldsB[buf ^ 1], 1u, K_DIM, BN, K_DIM, BK64, BN);
      }
#else
      for (int i = tid; i < BM * BK64 / 8; i += 256) {
        int r = (i * 8) / BK64, k = (i * 8) % BK64;
        reinterpret_cast<uint4*>(&sA[buf ^ 1][0])[i] =
            *reinterpret_cast<const uint4*>((const __bf16*)xbase + (size_t)r * K_DIM + k0 + k);
      }
      for (int i = tid; i < BN * BK64 / 8; i += 256) {
        int r = (i * 8) / BK64, k = (i * 8) % BK64;
        reinterpret_cast<uint4*>(&sB[buf ^ 1][0])[i] =
            *reinterpret_cast<const uint4*>((const __bf16*)wbase + (size_t)r * K_DIM + k0 + k);
      }
#endif
    }

    // Two K=32 sub-steps per slab -> 16 WMMAs per barrier.
#pragma unroll
    for (int kk = 0; kk < 2; ++kk) {
      const int kbase = kk * BK32;

      // A frags: 16x32 bf16 layout; lane<16: K {0..7,16..23}, lane>=16: +8.
      v16bf afrag[2];
#pragma unroll
      for (int mt = 0; mt < 2; ++mt) {
        const int row = waveM * 32 + mt * 16 + lrow;
        v16bf a;
        *reinterpret_cast<v8bf*>(&a) =
            *reinterpret_cast<const v8bf*>(&sA[buf][row * BK64 + kbase + halfsel * 8]);
        *(reinterpret_cast<v8bf*>(&a) + 1) =
            *reinterpret_cast<const v8bf*>(&sA[buf][row * BK64 + kbase + 16 + halfsel * 8]);
        afrag[mt] = a;
      }

      // B frags: lane holds one N-column; lane<16: K {0..7,16..23}, lane>=16: +8.
      v16bf bfrag[4];
#pragma unroll
      for (int nt = 0; nt < 4; ++nt) {
        const int col = waveN * 64 + nt * 16 + lrow;
        *reinterpret_cast<v8bf*>(&bfrag[nt]) =
            *reinterpret_cast<const v8bf*>(&sB[buf][col * BK64 + kbase + halfsel * 8]);
        *(reinterpret_cast<v8bf*>(&bfrag[nt]) + 1) =
            *reinterpret_cast<const v8bf*>(&sB[buf][col * BK64 + kbase + 16 + halfsel * 8]);
      }

#pragma unroll
      for (int mt = 0; mt < 2; ++mt)
#pragma unroll
        for (int nt = 0; nt < 4; ++nt)
          acc[mt][nt] = __builtin_amdgcn_wmma_f32_16x16x32_bf16(
              false, afrag[mt], false, bfrag[nt], (short)0, acc[mt][nt], false, false);
    }

#if USE_TDM
    if (wave == 0) __builtin_amdgcn_s_wait_tensorcnt(0);
#endif
    __syncthreads();
  }

  const float scale = input_scale[0] * weight_scale[0];
  float sumabs = 0.0f;
#pragma unroll
  for (int mt = 0; mt < 2; ++mt)
#pragma unroll
    for (int nt = 0; nt < 4; ++nt)
#pragma unroll
      for (int e = 0; e < 8; ++e) {
        const int grow = blockM + waveM * 32 + mt * 16 + e + halfsel * 8;
        const int gcol = blockN + waveN * 64 + nt * 16 + lrow;
        const float v = acc[mt][nt][e] * scale;
        out[(size_t)grow * N_DIM + gcol] = v;
        sumabs += fabsf(v);
      }

  __syncthreads();
  float* red = reinterpret_cast<float*>(&sA[0][0]);
  red[tid] = sumabs;
  __syncthreads();
#pragma unroll
  for (int s = 128; s > 0; s >>= 1) {
    if (tid < s) red[tid] += red[tid + s];
    __syncthreads();
  }
  if (tid == 0) partials[blockIdx.y * gridDim.x + blockIdx.x] = red[0];
}

// ---------------- Fallback: f32-A on-the-fly convert GEMM (round-1, BK=32) ----
__global__ __launch_bounds__(256) void bitnet_gemm_f32x(const float* __restrict__ x,
                                                        const float* __restrict__ input_scale,
                                                        const __bf16* __restrict__ wbf,
                                                        const float* __restrict__ weight_scale,
                                                        float* __restrict__ out,
                                                        float* __restrict__ partials) {
  __shared__ __align__(16) float  sA[2][BM * BK32];
  __shared__ __align__(16) __bf16 sB[2][BN * BK32];

  const int tid     = threadIdx.x;
  const int lane    = tid & 31;
  const int wave    = __builtin_amdgcn_readfirstlane(tid >> 5);
  const int waveM   = wave & 3;
  const int waveN   = wave >> 2;
  const int lrow    = lane & 15;
  const int halfsel = lane >> 4;

  const int blockM = blockIdx.y * BM;
  const int blockN = blockIdx.x * BN;

  v8f zero = {0.f, 0.f, 0.f, 0.f, 0.f, 0.f, 0.f, 0.f};
  v8f acc[2][4];
#pragma unroll
  for (int mt = 0; mt < 2; ++mt)
#pragma unroll
    for (int nt = 0; nt < 4; ++nt) acc[mt][nt] = zero;

  const unsigned long long xbase = (unsigned long long)(const void*)(x + (size_t)blockM * K_DIM);
  const unsigned long long wbase = (unsigned long long)(const void*)(wbf + (size_t)blockN * K_DIM);

#if USE_TDM
  const unsigned ldsA[2] = { (unsigned)(unsigned long long)(uintptr_t)&sA[0][0],
                             (unsigned)(unsigned long long)(uintptr_t)&sA[1][0] };
  const unsigned ldsB[2] = { (unsigned)(unsigned long long)(uintptr_t)&sB[0][0],
                             (unsigned)(unsigned long long)(uintptr_t)&sB[1][0] };
  if (wave == 0) {
    tdm_load_2d(xbase, ldsA[0], 2u, K_DIM, BM, K_DIM, BK32, BM);
    tdm_load_2d(wbase, ldsB[0], 1u, K_DIM, BN, K_DIM, BK32, BN);
    __builtin_amdgcn_s_wait_tensorcnt(0);
  }
  __syncthreads();
#else
  for (int i = tid; i < BM * BK32 / 4; i += 256) {
    int r = (i * 4) / BK32, k = (i * 4) % BK32;
    reinterpret_cast<float4*>(&sA[0][0])[i] =
        *reinterpret_cast<const float4*>((const float*)xbase + (size_t)r * K_DIM + k);
  }
  for (int i = tid; i < BN * BK32 / 8; i += 256) {
    int r = (i * 8) / BK32, k = (i * 8) % BK32;
    reinterpret_cast<uint4*>(&sB[0][0])[i] =
        *reinterpret_cast<const uint4*>((const __bf16*)wbase + (size_t)r * K_DIM + k);
  }
  __syncthreads();
#endif

#pragma unroll 1
  for (int kt = 0; kt < K_DIM / BK32; ++kt) {
    const int buf = kt & 1;

    if (kt + 1 < K_DIM / BK32) {
      const unsigned long long k0 = (unsigned long long)(kt + 1) * BK32;
#if USE_TDM
      if (wave == 0) {
        tdm_load_2d(xbase + k0 * 4, ldsA[buf ^ 1], 2u, K_DIM, BM, K_DIM, BK32, BM);
        tdm_load_2d(wbase + k0 * 2, ldsB[buf ^ 1], 1u, K_DIM, BN, K_DIM, BK32, BN);
      }
#else
      for (int i = tid; i < BM * BK32 / 4; i += 256) {
        int r = (i * 4) / BK32, k = (i * 4) % BK32;
        reinterpret_cast<float4*>(&sA[buf ^ 1][0])[i] =
            *reinterpret_cast<const float4*>((const float*)xbase + (size_t)r * K_DIM + k0 + k);
      }
      for (int i = tid; i < BN * BK32 / 8; i += 256) {
        int r = (i * 8) / BK32, k = (i * 8) % BK32;
        reinterpret_cast<uint4*>(&sB[buf ^ 1][0])[i] =
            *reinterpret_cast<const uint4*>((const __bf16*)wbase + (size_t)r * K_DIM + k0 + k);
      }
#endif
    }

    v16bf afrag[2];
#pragma unroll
    for (int mt = 0; mt < 2; ++mt) {
      const int row = waveM * 32 + mt * 16 + lrow;
      const float* fr = &sA[buf][row * BK32];
      v16bf a;
      __bf16* ap = reinterpret_cast<__bf16*>(&a);
#pragma unroll
      for (int i = 0; i < 4; ++i) {
        const int koff = ((i & 1) ? 4 : 0) + ((i >> 1) ? 16 : 0) + halfsel * 8;
        float4 f = *reinterpret_cast<const float4*>(fr + koff);
        ap[i * 4 + 0] = (__bf16)f.x; ap[i * 4 + 1] = (__bf16)f.y;
        ap[i * 4 + 2] = (__bf16)f.z; ap[i * 4 + 3] = (__bf16)f.w;
      }
      afrag[mt] = a;
    }

    v16bf bfrag[4];
#pragma unroll
    for (int nt = 0; nt < 4; ++nt) {
      const int col = waveN * 64 + nt * 16 + lrow;
      *reinterpret_cast<v8bf*>(&bfrag[nt]) =
          *reinterpret_cast<const v8bf*>(&sB[buf][col * BK32 + halfsel * 8]);
      *(reinterpret_cast<v8bf*>(&bfrag[nt]) + 1) =
          *reinterpret_cast<const v8bf*>(&sB[buf][col * BK32 + 16 + halfsel * 8]);
    }

#pragma unroll
    for (int mt = 0; mt < 2; ++mt)
#pragma unroll
      for (int nt = 0; nt < 4; ++nt)
        acc[mt][nt] = __builtin_amdgcn_wmma_f32_16x16x32_bf16(
            false, afrag[mt], false, bfrag[nt], (short)0, acc[mt][nt], false, false);

#if USE_TDM
    if (wave == 0) __builtin_amdgcn_s_wait_tensorcnt(0);
#endif
    __syncthreads();
  }

  const float scale = input_scale[0] * weight_scale[0];
  float sumabs = 0.0f;
#pragma unroll
  for (int mt = 0; mt < 2; ++mt)
#pragma unroll
    for (int nt = 0; nt < 4; ++nt)
#pragma unroll
      for (int e = 0; e < 8; ++e) {
        const int grow = blockM + waveM * 32 + mt * 16 + e + halfsel * 8;
        const int gcol = blockN + waveN * 64 + nt * 16 + lrow;
        const float v = acc[mt][nt][e] * scale;
        out[(size_t)grow * N_DIM + gcol] = v;
        sumabs += fabsf(v);
      }

  __syncthreads();
  float* red = &sA[0][0];
  red[tid] = sumabs;
  __syncthreads();
#pragma unroll
  for (int s = 128; s > 0; s >>= 1) {
    if (tid < s) red[tid] += red[tid + s];
    __syncthreads();
  }
  if (tid == 0) partials[blockIdx.y * gridDim.x + blockIdx.x] = red[0];
}

// ---------------- absmean scale reduce + quantize -----------------------------
__global__ __launch_bounds__(256) void bitnet_scale(const float* __restrict__ partials, int n,
                                                    float* __restrict__ scale_ws,
                                                    float* __restrict__ out_scale_slot,
                                                    long long numel) {
  __shared__ float red[256];
  float s = 0.0f;
  for (int i = threadIdx.x; i < n; i += 256) s += partials[i];
  red[threadIdx.x] = s;
  __syncthreads();
#pragma unroll
  for (int st = 128; st > 0; st >>= 1) {
    if ((int)threadIdx.x < st) red[threadIdx.x] += red[threadIdx.x + st];
    __syncthreads();
  }
  if (threadIdx.x == 0) {
    float sc = fmaxf(red[0] / (float)numel, 1e-5f);
    scale_ws[0] = sc;
    out_scale_slot[0] = sc;
  }
}

__global__ __launch_bounds__(256) void bitnet_quant(float* __restrict__ out,
                                                    const float* __restrict__ scale_ws,
                                                    long long n4) {
  long long i = (long long)blockIdx.x * 256 + threadIdx.x;
  if (i < n4) {
    const float inv = 1.0f / scale_ws[0];
    float4 v = reinterpret_cast<float4*>(out)[i];
    v.x = fminf(1.f, fmaxf(-1.f, rintf(v.x * inv)));
    v.y = fminf(1.f, fmaxf(-1.f, rintf(v.y * inv)));
    v.z = fminf(1.f, fmaxf(-1.f, rintf(v.z * inv)));
    v.w = fminf(1.f, fmaxf(-1.f, rintf(v.w * inv)));
    reinterpret_cast<float4*>(out)[i] = v;
  }
}

extern "C" void kernel_launch(void* const* d_in, const int* in_sizes, int n_in,
                              void* d_out, int out_size, void* d_ws, size_t ws_size,
                              hipStream_t stream) {
  const float* x           = (const float*)d_in[0];
  const float* input_scale = (const float*)d_in[1];
  const float* qw          = (const float*)d_in[2];
  const float* w_scale     = (const float*)d_in[3];
  float* out               = (float*)d_out;

  const int M = in_sizes[0] / K_DIM;                 // 16384
  const long long numel = (long long)M * N_DIM;      // 33,554,432
  const int gx = N_DIM / BN, gy = M / BM;            // 16 x 128 blocks
  const size_t xbytes = (size_t)numel * 0 + (size_t)in_sizes[0] * sizeof(__bf16); // 64 MB
  const size_t wbytes = (size_t)N_DIM * K_DIM * sizeof(__bf16);                   // 8 MB
  const size_t pbytes = (size_t)gx * gy * sizeof(float) + sizeof(float);

  const int wn4 = (N_DIM * K_DIM) / 4;
  const long long xn4 = (long long)in_sizes[0] / 4;
  dim3 grid(gx, gy);

  if (ws_size >= xbytes + wbytes + pbytes) {
    // Fast path: pre-convert x and w to bf16; all-bf16 GEMM with BK=64 TDM slabs.
    __bf16* xbf = (__bf16*)d_ws;
    __bf16* wbf = (__bf16*)((char*)d_ws + xbytes);
    float* partials = (float*)((char*)d_ws + xbytes + wbytes);
    float* scale_ws = partials + (size_t)gx * gy;

    bitnet_cvt<<<(unsigned)((xn4 + 255) / 256), 256, 0, stream>>>(x, xbf, (int)xn4);
    bitnet_cvt<<<(wn4 + 255) / 256, 256, 0, stream>>>(qw, wbf, wn4);
    bitnet_gemm_bf16<<<grid, 256, 0, stream>>>(xbf, input_scale, wbf, w_scale, out, partials);
    bitnet_scale<<<1, 256, 0, stream>>>(partials, gx * gy, scale_ws, out + numel, numel);
  } else {
    // Fallback: convert only w; GEMM converts x f32->bf16 at fragment build.
    __bf16* wbf = (__bf16*)d_ws;
    float* partials = (float*)((char*)d_ws + wbytes);
    float* scale_ws = partials + (size_t)gx * gy;

    bitnet_cvt<<<(wn4 + 255) / 256, 256, 0, stream>>>(qw, wbf, wn4);
    bitnet_gemm_f32x<<<grid, 256, 0, stream>>>(x, input_scale, wbf, w_scale, out, partials);
    bitnet_scale<<<1, 256, 0, stream>>>(partials, gx * gy, scale_ws, out + numel, numel);
  }

  // Final in-place ternary quantization of d_out (scale slot sits after partials
  // in both layouts; recompute its address the same way the chosen branch did).
  float* scale_ws = (ws_size >= xbytes + wbytes + pbytes)
      ? (float*)((char*)d_ws + xbytes + wbytes) + (size_t)gx * gy
      : (float*)((char*)d_ws + wbytes) + (size_t)gx * gy;
  const long long n4 = numel / 4;
  bitnet_quant<<<(unsigned)((n4 + 255) / 256), 256, 0, stream>>>(out, scale_ws, n4);
}
